// PointConvLayer_79431125172409
// MI455X (gfx1250) — matrix operand order repaired and checked by reading, not compiled, not run on hardware
//
#include <hip/hip_runtime.h>

#define BDIM 8
#define NPTS 4096
#define NPOINTS 1024
#define NSAMP 32
#define DFEAT 64
#define MCOLS (BDIM * NPOINTS * NSAMP)   // 262144 columns for conv2 chains
#define M2COLS (BDIM * NPTS)             // 32768 columns for density MLP
#define MSCOLS (BDIM * NPOINTS)          // 8192 rows for final linear
#define EPSV 1e-5f

typedef __bf16 bf16_t;
typedef bf16_t v16bf __attribute__((ext_vector_type(16)));
typedef float  v8f   __attribute__((ext_vector_type(8)));

union AFrag { v16bf v; uint4 u[2]; };
union BFrag { v16bf v; uint4 u[2]; };

// ---------------------------------------------------------------------------
// 1) Kernel-density estimate: dens[b][i] = mean_j exp(-50*d2(i,j)) / 0.25
// ---------------------------------------------------------------------------
__global__ __launch_bounds__(256) void k_density(const float* __restrict__ xyz,
                                                 float* __restrict__ dens) {
  int b = blockIdx.y;
  int i = blockIdx.x * 256 + threadIdx.x;
  const float* xb = xyz + (size_t)b * NPTS * 3;
  float px = xb[i * 3 + 0], py = xb[i * 3 + 1], pz = xb[i * 3 + 2];
  __shared__ float sx[256], sy[256], sz[256];
  float acc = 0.f;
  for (int j0 = 0; j0 < NPTS; j0 += 256) {
    int j = j0 + threadIdx.x;
    __syncthreads();
    sx[threadIdx.x] = xb[j * 3 + 0];
    sy[threadIdx.x] = xb[j * 3 + 1];
    sz[threadIdx.x] = xb[j * 3 + 2];
    __syncthreads();
#pragma unroll 8
    for (int jj = 0; jj < 256; ++jj) {
      float dx = px - sx[jj], dy = py - sy[jj], dz = pz - sz[jj];
      acc += __expf(-50.f * (dx * dx + dy * dy + dz * dz));
    }
  }
  dens[(size_t)b * NPTS + i] = acc * (1.f / (NPTS * 0.25f));
}

// ---------------------------------------------------------------------------
// Small channel-wise conv (Y[co][m] = sum_ci W[co][ci] * X[ci][m]); bias
// omitted (cancels under batch-norm).
// ---------------------------------------------------------------------------
__global__ void k_small_conv(const float* __restrict__ X, const float* __restrict__ W,
                             float* __restrict__ Y, int Cin, int Cout, int M) {
  int m = blockIdx.x * blockDim.x + threadIdx.x;
  if (m >= M) return;
  for (int co = 0; co < Cout; ++co) {
    float acc = 0.f;
    for (int ci = 0; ci < Cin; ++ci)
      acc += W[co * Cin + ci] * X[(size_t)ci * M + m];
    Y[(size_t)co * M + m] = acc;
  }
}

// ---------------------------------------------------------------------------
// Deterministic two-stage per-channel sum / sumsq reduction for batch-norm.
// ---------------------------------------------------------------------------
__global__ __launch_bounds__(256) void k_colstats(const float* __restrict__ Y, int M,
                                                  const float* __restrict__ bias,
                                                  int relu_first,
                                                  float* __restrict__ partial) {
  int c = blockIdx.y;
  int nb = gridDim.x;
  float bc = bias ? bias[c] : 0.f;
  float s = 0.f, s2 = 0.f;
  for (int m = blockIdx.x * 256 + threadIdx.x; m < M; m += nb * 256) {
    float v = Y[(size_t)c * M + m] + bc;
    if (relu_first) v = fmaxf(v, 0.f);
    s += v;
    s2 += v * v;
  }
  __shared__ float sh1[256], sh2[256];
  sh1[threadIdx.x] = s;
  sh2[threadIdx.x] = s2;
  __syncthreads();
  for (int off = 128; off > 0; off >>= 1) {
    if ((int)threadIdx.x < off) {
      sh1[threadIdx.x] += sh1[threadIdx.x + off];
      sh2[threadIdx.x] += sh2[threadIdx.x + off];
    }
    __syncthreads();
  }
  if (threadIdx.x == 0) {
    partial[((size_t)c * nb + blockIdx.x) * 2 + 0] = sh1[0];
    partial[((size_t)c * nb + blockIdx.x) * 2 + 1] = sh2[0];
  }
}

__global__ void k_redstats(const float* __restrict__ partial, int nb, int C,
                           float* __restrict__ sums) {
  int c = blockIdx.x * blockDim.x + threadIdx.x;
  if (c >= C) return;
  float s = 0.f, s2 = 0.f;
  for (int i = 0; i < nb; ++i) {
    s  += partial[((size_t)c * nb + i) * 2 + 0];
    s2 += partial[((size_t)c * nb + i) * 2 + 1];
  }
  sums[2 * c] = s;
  sums[2 * c + 1] = s2;
}

// Batch-norm + relu, channel-major output (feeds A-side of agg WMMA).
template <typename OUT>
__global__ void k_bnrelu(const float* __restrict__ Y, int M,
                         const float* __restrict__ g, const float* __restrict__ t,
                         const float* __restrict__ sums, OUT* __restrict__ out) {
  int c = blockIdx.y;
  float invM = 1.f / (float)M;
  float mean = sums[2 * c] * invM;
  float var  = sums[2 * c + 1] * invM - mean * mean;
  float inv = rsqrtf(var + EPSV);
  float gg = g[c], tt = t[c];
  for (int m = blockIdx.x * blockDim.x + threadIdx.x; m < M;
       m += gridDim.x * blockDim.x) {
    float v = gg * (Y[(size_t)c * M + m] - mean) * inv + tt;
    v = fmaxf(v, 0.f);
    out[(size_t)c * M + m] = (OUT)v;
  }
}

// Batch-norm + relu with TRANSPOSED output XT[m][C]: contiguous per-lane rows
// so the next WMMA GEMM's B fragment is two b128 loads (no u16 gathers).
template <typename OUT>
__global__ void k_bnrelu_T(const float* __restrict__ Y, int M, int C,
                           const float* __restrict__ g, const float* __restrict__ t,
                           const float* __restrict__ sums, OUT* __restrict__ outT) {
  int m = blockIdx.x * blockDim.x + threadIdx.x;
  if (m >= M) return;
  float invM = 1.f / (float)M;
  for (int c = 0; c < C; ++c) {
    float mean = sums[2 * c] * invM;
    float var  = sums[2 * c + 1] * invM - mean * mean;
    float inv = rsqrtf(var + EPSV);
    float v = g[c] * (Y[(size_t)c * M + m] - mean) * inv + t[c];
    outT[(size_t)m * C + c] = (OUT)fmaxf(v, 0.f);
  }
}

// ---------------------------------------------------------------------------
// Farthest point sampling: one 1024-thread block per batch, 1024 sequential
// steps with an LDS argmax (first-max tie-break like jnp.argmax).
// ---------------------------------------------------------------------------
__global__ __launch_bounds__(1024) void k_fps(const float* __restrict__ xyz,
                                              int* __restrict__ fps_idx,
                                              float* __restrict__ new_xyz) {
  int b = blockIdx.x, t = threadIdx.x;
  const float* xb = xyz + (size_t)b * NPTS * 3;
  float lx[4], ly[4], lz[4], md[4];
#pragma unroll
  for (int q = 0; q < 4; ++q) {
    int p = t + q * 1024;
    lx[q] = xb[p * 3 + 0];
    ly[q] = xb[p * 3 + 1];
    lz[q] = xb[p * 3 + 2];
    md[q] = 1e10f;
  }
  __shared__ float sdist[1024];
  __shared__ int sidx[1024];
  __shared__ int sfps[1024];
  __shared__ float cpt[3];
  __shared__ int sfar;
  int far = 0;
  for (int it = 0; it < NPOINTS; ++it) {
    if (t == 0) {
      sfps[it] = far;
      fps_idx[(size_t)b * NPOINTS + it] = far;
      cpt[0] = xb[far * 3 + 0];
      cpt[1] = xb[far * 3 + 1];
      cpt[2] = xb[far * 3 + 2];
    }
    __syncthreads();
    float cx = cpt[0], cy = cpt[1], cz = cpt[2];
    float best = -1.f;
    int bidx = 0;
#pragma unroll
    for (int q = 0; q < 4; ++q) {
      float dx = lx[q] - cx, dy = ly[q] - cy, dz = lz[q] - cz;
      float d = dx * dx + dy * dy + dz * dz;
      md[q] = fminf(md[q], d);
      if (md[q] > best) { best = md[q]; bidx = t + q * 1024; }
    }
    sdist[t] = best;
    sidx[t] = bidx;
    __syncthreads();
    for (int off = 512; off > 0; off >>= 1) {
      if (t < off) {
        float o = sdist[t + off];
        int oi = sidx[t + off];
        if (o > sdist[t] || (o == sdist[t] && oi < sidx[t])) {
          sdist[t] = o;
          sidx[t] = oi;
        }
      }
      __syncthreads();
    }
    if (t == 0) sfar = sidx[0];
    __syncthreads();
    far = sfar;
  }
  int id = sfps[t];
  new_xyz[((size_t)b * NPOINTS + t) * 3 + 0] = xb[id * 3 + 0];
  new_xyz[((size_t)b * NPOINTS + t) * 3 + 1] = xb[id * 3 + 1];
  new_xyz[((size_t)b * NPOINTS + t) * 3 + 2] = xb[id * 3 + 2];
}

// ---------------------------------------------------------------------------
// KNN: one block per query, 32 min-extractions over LDS-resident distances.
// ---------------------------------------------------------------------------
__global__ __launch_bounds__(256) void k_knn(const float* __restrict__ xyz,
                                             const float* __restrict__ new_xyz,
                                             int* __restrict__ knn_idx) {
  int bs = blockIdx.x;
  int b = bs >> 10;
  int t = threadIdx.x;
  const float* xb = xyz + (size_t)b * NPTS * 3;
  float qx = new_xyz[(size_t)bs * 3 + 0];
  float qy = new_xyz[(size_t)bs * 3 + 1];
  float qz = new_xyz[(size_t)bs * 3 + 2];
  __shared__ float dist[NPTS];
  for (int j = t; j < NPTS; j += 256) {
    float dx = xb[j * 3 + 0] - qx, dy = xb[j * 3 + 1] - qy, dz = xb[j * 3 + 2] - qz;
    dist[j] = dx * dx + dy * dy + dz * dz;
  }
  __shared__ float rmin[256];
  __shared__ int ridx[256];
  for (int k = 0; k < NSAMP; ++k) {
    __syncthreads();
    float best = 3.0e38f;
    int bi = NPTS;
    for (int j = t; j < NPTS; j += 256) {
      float d = dist[j];
      if (d < best) { best = d; bi = j; }
    }
    rmin[t] = best;
    ridx[t] = bi;
    __syncthreads();
    for (int off = 128; off > 0; off >>= 1) {
      if (t < off) {
        float o = rmin[t + off];
        int oi = ridx[t + off];
        if (o < rmin[t] || (o == rmin[t] && oi < ridx[t])) {
          rmin[t] = o;
          ridx[t] = oi;
        }
      }
      __syncthreads();
    }
    if (t == 0) {
      knn_idx[(size_t)bs * NSAMP + k] = ridx[0];
      dist[ridx[0]] = 3.0e38f;
    }
  }
}

// ---------------------------------------------------------------------------
// Gather: build mparams input X0T[m][96] (bf16, cols 67..95 zero pad; each
// thread writes one contiguous 192B row) and wparams input XW (fp32, 3xM).
// Column order m = ((b*1024+s)*32 + k) so each (b,s) slice is contiguous.
// ---------------------------------------------------------------------------
__global__ void k_gather(const float* __restrict__ xyz, const float* __restrict__ points,
                         const float* __restrict__ new_xyz, const float* __restrict__ dscale,
                         const int* __restrict__ knn_idx,
                         bf16_t* __restrict__ X0T, float* __restrict__ XW) {
  int m = blockIdx.x * blockDim.x + threadIdx.x;
  if (m >= MCOLS) return;
  int bs = m >> 5;
  int b = bs >> 10;
  int j = knn_idx[m];
  float nx = new_xyz[(size_t)bs * 3 + 0];
  float ny = new_xyz[(size_t)bs * 3 + 1];
  float nz = new_xyz[(size_t)bs * 3 + 2];
  const float* xp = xyz + ((size_t)b * NPTS + j) * 3;
  float gx = xp[0] - nx, gy = xp[1] - ny, gz = xp[2] - nz;
  bf16_t* row = X0T + (size_t)m * 96;
  row[0] = (bf16_t)gx;
  row[1] = (bf16_t)gy;
  row[2] = (bf16_t)gz;
  const float* pt = points + ((size_t)b * NPTS + j) * DFEAT;
  for (int c = 0; c < DFEAT; ++c) row[3 + c] = (bf16_t)pt[c];
  for (int c = 3 + DFEAT; c < 96; ++c) row[c] = (bf16_t)0.f;
  float dv = dscale[(size_t)b * NPTS + j];
  XW[(size_t)0 * MCOLS + m] = gx * dv;
  XW[(size_t)1 * MCOLS + m] = gy * dv;
  XW[(size_t)2 * MCOLS + m] = gz * dv;
}

// Weight conversions to bf16 (with K zero-padding to a multiple of 32).
__global__ void k_convw(const float* __restrict__ src, bf16_t* __restrict__ dst,
                        int Cout, int Cin, int Kpad) {
  int i = blockIdx.x * blockDim.x + threadIdx.x;
  if (i >= Cout * Kpad) return;
  int c = i / Kpad, k = i % Kpad;
  dst[i] = (k < Cin) ? (bf16_t)src[c * Cin + k] : (bf16_t)0.f;
}

__global__ void k_convlwT(const float* __restrict__ lw, bf16_t* __restrict__ lwT) {
  int i = blockIdx.x * blockDim.x + threadIdx.x;
  if (i >= 128 * 2048) return;
  int o = i >> 11, k = i & 2047;
  lwT[i] = (bf16_t)lw[(size_t)k * 128 + o];
}

// ---------------------------------------------------------------------------
// bf16 WMMA GEMM: Y[Cout][M] = W[Cout][K] * XT[M][K]^T, f32 accumulate.
// One wave per 16x16 tile, K stepped by 32 (v_wmma_f32_16x16x32_bf16).
// Both fragments are pairs of contiguous b128 loads (ISA 7.12.2 layouts):
//   A lane: row c0+l, K chunks {kb+half*8.. , kb+16+half*8..}
//   B lane: row m0+l of XT, K chunks {kb+half*8.. , kb+16+half*8..}
// ---------------------------------------------------------------------------
__global__ __launch_bounds__(128) void k_gemm_wmma(const bf16_t* __restrict__ W,
                                                   const bf16_t* __restrict__ XT,
                                                   float* __restrict__ Y,
                                                   int K, int M) {
  int lane = threadIdx.x & 31;
  int wv = threadIdx.x >> 5;
  int m0 = (blockIdx.x * 4 + wv) * 16;
  int c0 = blockIdx.y * 16;
  if (m0 >= M) return;  // wave-uniform
  int half = lane >> 4;
  int l = lane & 15;
  const bf16_t* arow = W + (size_t)(c0 + l) * K + half * 8;
  const bf16_t* brow = XT + (size_t)(m0 + l) * K + half * 8;
  v8f acc = {};
  for (int kb = 0; kb < K; kb += 32) {
    AFrag a;
    a.u[0] = *(const uint4*)(arow + kb);
    a.u[1] = *(const uint4*)(arow + kb + 16);
    BFrag bm;
    bm.u[0] = *(const uint4*)(brow + kb);
    bm.u[1] = *(const uint4*)(brow + kb + 16);
    acc = __builtin_amdgcn_wmma_f32_16x16x32_bf16(false, a.v, false, bm.v,
                                                  (short)0, acc, false, false);
  }
#pragma unroll
  for (int r = 0; r < 8; ++r)
    Y[(size_t)(c0 + half * 8 + r) * M + (m0 + l)] = acc[r];
}

// ---------------------------------------------------------------------------
// agg einsum: per (b,s), (128x32)F * (32x16)G^T via 8 WMMA tiles (K=32 ==
// exactly one v_wmma_f32_16x16x32_bf16). F and GX are channel-major so the
// sample axis (K) is contiguous for both fragments. Output agg2[bs][c*16+w]
// is row-major (b,s)x2048 -- both coalesced here and exactly the XT layout
// the final WMMA GEMM consumes.
// ---------------------------------------------------------------------------
__global__ __launch_bounds__(256) void k_agg_wmma(const bf16_t* __restrict__ F,
                                                  const bf16_t* __restrict__ GX,
                                                  bf16_t* __restrict__ agg2) {
  int bs = blockIdx.x;
  int lane = threadIdx.x & 31;
  int ctile = threadIdx.x >> 5;  // 0..7
  int half = lane >> 4;
  int l = lane & 15;
  int m0 = bs * NSAMP;
  int c0 = ctile * 16;
  AFrag a;
  const bf16_t* arow = F + (size_t)(c0 + l) * MCOLS + m0 + half * 8;
  a.u[0] = *(const uint4*)(arow);
  a.u[1] = *(const uint4*)(arow + 16);
  BFrag bm;
  const bf16_t* g = GX + (size_t)l * MCOLS + m0 + half * 8;
  bm.u[0] = *(const uint4*)(g);
  bm.u[1] = *(const uint4*)(g + 16);
  v8f acc = {};
  acc = __builtin_amdgcn_wmma_f32_16x16x32_bf16(false, a.v, false, bm.v,
                                                (short)0, acc, false, false);
#pragma unroll
  for (int r = 0; r < 8; ++r) {
    int c = c0 + half * 8 + r;
    agg2[(size_t)bs * 2048 + c * 16 + l] = (bf16_t)acc[r];
  }
}

// Final: v = relu(Yf + lb); out = lg*(v-mean)*rsqrt(var+eps)+lt, transposed
// to the reference (B, NPOINT, 128) layout.
__global__ void k_final(const float* __restrict__ Yf, const float* __restrict__ lb,
                        const float* __restrict__ lg, const float* __restrict__ lt,
                        const float* __restrict__ sums, float* __restrict__ out) {
  int c = blockIdx.y;
  int m = blockIdx.x * blockDim.x + threadIdx.x;
  if (m >= MSCOLS) return;
  float invM = 1.f / (float)MSCOLS;
  float mean = sums[2 * c] * invM;
  float var = sums[2 * c + 1] * invM - mean * mean;
  float inv = rsqrtf(var + EPSV);
  float v = fmaxf(Yf[(size_t)c * MSCOLS + m] + lb[c], 0.f);
  out[(size_t)m * 128 + c] = lg[c] * (v - mean) * inv + lt[c];
}

// ---------------------------------------------------------------------------
extern "C" void kernel_launch(void* const* d_in, const int* in_sizes, int n_in,
                              void* d_out, int out_size, void* d_ws, size_t ws_size,
                              hipStream_t stream) {
  (void)in_sizes; (void)n_in; (void)out_size; (void)ws_size;
  const float* xyz = (const float*)d_in[0];
  const float* points = (const float*)d_in[1];
  const float *dw[3], *dg[3], *dt[3];
  for (int i = 0; i < 3; ++i) {
    dw[i] = (const float*)d_in[2 + 4 * i];
    dg[i] = (const float*)d_in[2 + 4 * i + 2];
    dt[i] = (const float*)d_in[2 + 4 * i + 3];
  }
  const float *ww[3], *wg[3], *wt[3];
  for (int i = 0; i < 3; ++i) {
    ww[i] = (const float*)d_in[14 + 4 * i];
    wg[i] = (const float*)d_in[14 + 4 * i + 2];
    wt[i] = (const float*)d_in[14 + 4 * i + 3];
  }
  const float *mw[2], *mg[2], *mt[2];
  for (int i = 0; i < 2; ++i) {
    mw[i] = (const float*)d_in[26 + 4 * i];
    mg[i] = (const float*)d_in[26 + 4 * i + 2];
    mt[i] = (const float*)d_in[26 + 4 * i + 3];
  }
  const float* lw = (const float*)d_in[34];
  const float* lb = (const float*)d_in[35];
  const float* lg = (const float*)d_in[36];
  const float* lt = (const float*)d_in[37];
  float* outp = (float*)d_out;

  char* ws = (char*)d_ws;
  size_t off = 0;
  auto alloc = [&](size_t bytes) -> void* {
    off = (off + 255) & ~(size_t)255;
    void* p = ws + off;
    off += bytes;
    return p;
  };
  float* dens    = (float*)alloc((size_t)M2COLS * 4);
  float* s0      = (float*)alloc((size_t)8 * M2COLS * 4);
  float* s1      = (float*)alloc((size_t)8 * M2COLS * 4);
  float* dscale  = (float*)alloc((size_t)M2COLS * 4);
  int*   fpsIdx  = (int*)alloc((size_t)MSCOLS * 4);
  float* nxyz    = (float*)alloc((size_t)MSCOLS * 3 * 4);
  int*   knnIdx  = (int*)alloc((size_t)MCOLS * 4);
  float* partial = (float*)alloc((size_t)128 * 64 * 2 * 4);
  float* sums    = (float*)alloc((size_t)2 * 128 * 4);
  bf16_t* Wm1    = (bf16_t*)alloc((size_t)64 * 96 * 2);
  bf16_t* Wm2    = (bf16_t*)alloc((size_t)128 * 64 * 2);
  bf16_t* lwT    = (bf16_t*)alloc((size_t)128 * 2048 * 2);
  bf16_t* bufB   = (bf16_t*)alloc((size_t)128 * MCOLS * 2);  // X0T -> X1T -> F
  float*  bufY   = (float*)alloc((size_t)128 * MCOLS * 4);   // Y1 / Y2
  float*  w0     = (float*)alloc((size_t)16 * MCOLS * 4);
  float*  w1     = (float*)alloc((size_t)16 * MCOLS * 4);
  bf16_t* gxb    = (bf16_t*)alloc((size_t)16 * MCOLS * 2);
  bf16_t* agg2   = (bf16_t*)alloc((size_t)2048 * MSCOLS * 2);
  float*  Yf     = (float*)alloc((size_t)128 * MSCOLS * 4);

  dim3 b256(256);
  const int nb2 = 16, nbM = 64, nbS = 8;

  // 1) density
  k_density<<<dim3(NPTS / 256, BDIM), b256, 0, stream>>>(xyz, dens);

  // 2) density MLP chain (1->8->8->1), BN over (B,N) per channel
  k_small_conv<<<dim3(M2COLS / 256), b256, 0, stream>>>(dens, dw[0], s0, 1, 8, M2COLS);
  k_colstats<<<dim3(nb2, 8), b256, 0, stream>>>(s0, M2COLS, nullptr, 0, partial);
  k_redstats<<<dim3(1), dim3(128), 0, stream>>>(partial, nb2, 8, sums);
  k_bnrelu<float><<<dim3(nb2, 8), b256, 0, stream>>>(s0, M2COLS, dg[0], dt[0], sums, s1);
  k_small_conv<<<dim3(M2COLS / 256), b256, 0, stream>>>(s1, dw[1], s0, 8, 8, M2COLS);
  k_colstats<<<dim3(nb2, 8), b256, 0, stream>>>(s0, M2COLS, nullptr, 0, partial);
  k_redstats<<<dim3(1), dim3(128), 0, stream>>>(partial, nb2, 8, sums);
  k_bnrelu<float><<<dim3(nb2, 8), b256, 0, stream>>>(s0, M2COLS, dg[1], dt[1], sums, s1);
  k_small_conv<<<dim3(M2COLS / 256), b256, 0, stream>>>(s1, dw[2], s0, 8, 1, M2COLS);
  k_colstats<<<dim3(nb2, 1), b256, 0, stream>>>(s0, M2COLS, nullptr, 0, partial);
  k_redstats<<<dim3(1), dim3(128), 0, stream>>>(partial, nb2, 1, sums);
  k_bnrelu<float><<<dim3(nb2, 1), b256, 0, stream>>>(s0, M2COLS, dg[2], dt[2], sums, dscale);

  // 3) FPS + 4) KNN + 5) gather
  k_fps<<<dim3(BDIM), dim3(1024), 0, stream>>>(xyz, fpsIdx, nxyz);
  k_knn<<<dim3(MSCOLS), b256, 0, stream>>>(xyz, nxyz, knnIdx);
  bf16_t* X0T = bufB;
  k_gather<<<dim3(MCOLS / 256), b256, 0, stream>>>(xyz, points, nxyz, dscale, knnIdx, X0T, w0);

  // 6) wparams chain (3->8->8->16), fp32 VALU, last output to bf16 for WMMA agg
  k_small_conv<<<dim3(MCOLS / 256), b256, 0, stream>>>(w0, ww[0], w1, 3, 8, MCOLS);
  k_colstats<<<dim3(nbM, 8), b256, 0, stream>>>(w1, MCOLS, nullptr, 0, partial);
  k_redstats<<<dim3(1), dim3(128), 0, stream>>>(partial, nbM, 8, sums);
  k_bnrelu<float><<<dim3(nbM, 8), b256, 0, stream>>>(w1, MCOLS, wg[0], wt[0], sums, w0);
  k_small_conv<<<dim3(MCOLS / 256), b256, 0, stream>>>(w0, ww[1], w1, 8, 8, MCOLS);
  k_colstats<<<dim3(nbM, 8), b256, 0, stream>>>(w1, MCOLS, nullptr, 0, partial);
  k_redstats<<<dim3(1), dim3(128), 0, stream>>>(partial, nbM, 8, sums);
  k_bnrelu<float><<<dim3(nbM, 8), b256, 0, stream>>>(w1, MCOLS, wg[1], wt[1], sums, w0);
  k_small_conv<<<dim3(MCOLS / 256), b256, 0, stream>>>(w0, ww[2], w1, 8, 16, MCOLS);
  k_colstats<<<dim3(nbM, 16), b256, 0, stream>>>(w1, MCOLS, nullptr, 0, partial);
  k_redstats<<<dim3(1), dim3(128), 0, stream>>>(partial, nbM, 16, sums);
  k_bnrelu<bf16_t><<<dim3(nbM, 16), b256, 0, stream>>>(w1, MCOLS, wg[2], wt[2], sums, gxb);

  // 7) weight conversion for WMMA GEMMs
  k_convw<<<dim3((64 * 96 + 255) / 256), b256, 0, stream>>>(mw[0], Wm1, 64, 67, 96);
  k_convw<<<dim3((128 * 64 + 255) / 256), b256, 0, stream>>>(mw[1], Wm2, 128, 64, 64);
  k_convlwT<<<dim3((128 * 2048 + 255) / 256), b256, 0, stream>>>(lw, lwT);

  // 8) mparams chain via WMMA: [M][96] -> 64xM -> [M][64] -> 128xM
  k_gemm_wmma<<<dim3(MCOLS / 64, 4), dim3(128), 0, stream>>>(Wm1, X0T, bufY, 96, MCOLS);
  k_colstats<<<dim3(nbM, 64), b256, 0, stream>>>(bufY, MCOLS, nullptr, 0, partial);
  k_redstats<<<dim3(1), dim3(128), 0, stream>>>(partial, nbM, 64, sums);
  bf16_t* X1T = bufB;
  k_bnrelu_T<bf16_t><<<dim3(MCOLS / 256), b256, 0, stream>>>(bufY, MCOLS, 64, mg[0], mt[0], sums, X1T);
  k_gemm_wmma<<<dim3(MCOLS / 64, 8), dim3(128), 0, stream>>>(Wm2, X1T, bufY, 64, MCOLS);
  k_colstats<<<dim3(nbM, 128), b256, 0, stream>>>(bufY, MCOLS, nullptr, 0, partial);
  k_redstats<<<dim3(1), dim3(128), 0, stream>>>(partial, nbM, 128, sums);
  bf16_t* F = bufB;  // channel-major [128][M] for the agg A-side
  k_bnrelu<bf16_t><<<dim3(nbM, 128), b256, 0, stream>>>(bufY, MCOLS, mg[1], mt[1], sums, F);

  // 9) agg einsum via WMMA (8 tiles per (b,s)) -> agg2[8192][2048]
  k_agg_wmma<<<dim3(MSCOLS), dim3(256), 0, stream>>>(F, gxb, agg2);

  // 10) final linear (128x2048)x(2048x8192) via WMMA, then bias+relu+BN
  k_gemm_wmma<<<dim3(MSCOLS / 64, 8), dim3(128), 0, stream>>>(lwT, agg2, Yf, 2048, MSCOLS);
  k_colstats<<<dim3(nbS, 128), b256, 0, stream>>>(Yf, MSCOLS, lb, 1, partial);
  k_redstats<<<dim3(1), dim3(128), 0, stream>>>(partial, nbS, 128, sums);
  k_final<<<dim3(MSCOLS / 256, 128), b256, 0, stream>>>(Yf, lb, lg, lt, sums, outp);
}
